// Autoformer_use_to_cat_48490180772003
// MI455X (gfx1250) — compile-verified, matrix operand order checked by reading, not run on hardware
//
#include <hip/hip_runtime.h>
#include <cstdint>
#include <cstddef>

typedef __attribute__((ext_vector_type(8)))  _Float16 v8h;
typedef __attribute__((ext_vector_type(16))) _Float16 v16h;
typedef __attribute__((ext_vector_type(8)))  float    v8f;

#define BATCHN 64
#define SEQL   1536
#define DMODEL 256
#define DFF    1024
#define NFFT   4096
#define TOPK   7

__device__ __forceinline__ float gelu_f(float x) {
  return 0.5f * x * (1.0f + erff(x * 0.7071067811865475f));
}
__device__ __forceinline__ float2 cmul2(float2 a, float2 b) {
  return make_float2(a.x * b.x - a.y * b.y, a.x * b.y + a.y * b.x);
}

// ---------------------------------------------------------------------------
// GEMM: C[M,N] = A[M,K](f32) @ W[N,K]^T(f16) + bias ; compile-time epilogue
// EPI: 0 = bias; 1 = bias + exact GELU; 2 = bias + residual add (R[M,N])
// Block: 256 threads = 8 wave32, tile 128(M) x 64(N), K-step 32.
// B (weight) tile staged with CDNA5 async global->LDS copies (ASYNCcnt path);
// A tile converted f32->f16 through VGPRs.
// ---------------------------------------------------------------------------
template <int EPI>
__global__ void __launch_bounds__(256) gemm_wmma_kernel(
    const float* __restrict__ A, const _Float16* __restrict__ W,
    const float* __restrict__ bias, const float* __restrict__ R,
    float* __restrict__ C, int M, int N, int K)
{
  __shared__ __align__(16) _Float16 As[128][32];
  __shared__ __align__(16) _Float16 Bs[64][32];
  const int tid  = threadIdx.x;
  const int lane = tid & 31;
  const int wave = tid >> 5;
  const int mblk = blockIdx.x * 128;
  const int nblk = blockIdx.y * 64;
  const int wm   = (wave & 3) * 32;
  const int wn   = (wave >> 2) * 32;
  const int l16  = lane & 15;
  const int kh   = (lane < 16) ? 0 : 8;   // K-half select per 16-bit A layout

  v8f acc[2][2] = {};

  #pragma unroll 2
  for (int k0 = 0; k0 < K; k0 += 32) {
    { // B tile 64x32 halves: one async b128 copy per thread, LDS written by DMA
      int r = tid >> 2, c = (tid & 3) * 8;
      const _Float16* gp = W + (size_t)(nblk + r) * K + k0 + c;
      unsigned ldsoff = (unsigned)(uintptr_t)&Bs[r][c];   // addr[31:0] == LDS byte offset
      asm volatile("global_load_async_to_lds_b128 %0, %1, off"
                   :: "v"(ldsoff), "v"(gp) : "memory");
    }
    { // A tile 128x32: 2 threads/row, 16 halves each (4x float4 loads, cvt f32->f16)
      int r = tid >> 1, c = (tid & 1) * 16;
      const float4* src =
          reinterpret_cast<const float4*>(A + (size_t)(mblk + r) * K + k0 + c);
      #pragma unroll
      for (int i = 0; i < 4; ++i) {
        float4 v = src[i];
        As[r][c + 4 * i + 0] = (_Float16)v.x;
        As[r][c + 4 * i + 1] = (_Float16)v.y;
        As[r][c + 4 * i + 2] = (_Float16)v.z;
        As[r][c + 4 * i + 3] = (_Float16)v.w;
      }
    }
    asm volatile("s_wait_asynccnt 0x0" ::: "memory");  // drain this wave's async copies
    __syncthreads();
    #pragma unroll
    for (int mi = 0; mi < 2; ++mi) {
      int row = wm + mi * 16 + l16;
      v8h alo = *reinterpret_cast<const v8h*>(&As[row][kh]);
      v8h ahi = *reinterpret_cast<const v8h*>(&As[row][kh + 16]);
      v16h a  = __builtin_shufflevector(alo, ahi, 0,1,2,3,4,5,6,7,8,9,10,11,12,13,14,15);
      #pragma unroll
      for (int ni = 0; ni < 2; ++ni) {
        int col = wn + ni * 16 + l16;
        v8h blo = *reinterpret_cast<const v8h*>(&Bs[col][kh]);
        v8h bhi = *reinterpret_cast<const v8h*>(&Bs[col][kh + 16]);
        v16h b  = __builtin_shufflevector(blo, bhi, 0,1,2,3,4,5,6,7,8,9,10,11,12,13,14,15);
        acc[mi][ni] = __builtin_amdgcn_wmma_f32_16x16x32_f16(
            false, a, false, b, (short)0, acc[mi][ni], false, false);
      }
    }
    __syncthreads();
  }
  // C/D layout: VGPR g -> row = g + 8*(lane/16), col = lane%16
  const int rsel = (lane >> 4) * 8;
  #pragma unroll
  for (int mi = 0; mi < 2; ++mi) {
    #pragma unroll
    for (int ni = 0; ni < 2; ++ni) {
      const int colg = nblk + wn + ni * 16 + l16;
      const float bv = bias[colg];
      const size_t base = (size_t)(mblk + wm + mi * 16 + rsel) * N + colg;
      #pragma unroll
      for (int g = 0; g < 8; ++g) {
        float v = acc[mi][ni][g] + bv;
        if (EPI == 1) v = gelu_f(v);
        if (EPI == 2) v += R[base + (size_t)g * N];
        C[base + (size_t)g * N] = v;
      }
    }
  }
}

// ---------------------------------------------------------------------------
// f32 -> f16 weight conversion
// ---------------------------------------------------------------------------
__global__ void cvt_f16_kernel(const float* __restrict__ in,
                               _Float16* __restrict__ out, int n) {
  int i = blockIdx.x * 256 + threadIdx.x;
  if (i < n) out[i] = (_Float16)in[i];
}

// ---------------------------------------------------------------------------
// Token embedding: circular conv1d(7 -> 256, k=3).  grid (L, B), block 256 (=d)
// ---------------------------------------------------------------------------
__global__ void __launch_bounds__(256) embed_kernel(
    const float* __restrict__ xe, const float* __restrict__ ew,
    float* __restrict__ X)
{
  int t = blockIdx.x, bb = blockIdx.y, d = threadIdx.x;
  float s = 0.0f;
  #pragma unroll
  for (int j = 0; j < 3; ++j) {
    int tt = t + j - 1;
    tt = (tt < 0) ? tt + SEQL : (tt >= SEQL ? tt - SEQL : tt);
    const float* row = xe + ((size_t)bb * SEQL + tt) * 7;
    #pragma unroll
    for (int c = 0; c < 7; ++c) s += row[c] * ew[(d * 7 + c) * 3 + j];
  }
  X[((size_t)bb * SEQL + t) * DMODEL + d] = s;
}

// ---------------------------------------------------------------------------
// Packed FFT of q+ik (4096-pt, zero-padded from 1536), per-group partial
// cross-spectra:  pgrp[b,g,f] = sum_{d in group} Qhat[f] * conj(Khat[f])
// grid (16 groups, B), block 256.
// ---------------------------------------------------------------------------
__global__ void __launch_bounds__(256) fft_qk_kernel(
    const float* __restrict__ Q, const float* __restrict__ Kv,
    float* __restrict__ pgrp)
{
  __shared__ float2 buf[NFFT];   // 32 KB
  __shared__ float2 pac[NFFT];   // 32 KB
  int g = blockIdx.x, bb = blockIdx.y, tid = threadIdx.x;
  for (int j = tid; j < NFFT; j += 256) pac[j] = make_float2(0.f, 0.f);
  const size_t rb = (size_t)bb * SEQL;

  for (int dc = 0; dc < 16; ++dc) {
    int d = g * 16 + dc;
    __syncthreads();
    for (int j = tid; j < NFFT; j += 256) {     // bit-reversed load, zero-pad
      int src = __brev(j) >> 20;
      float2 z = make_float2(0.f, 0.f);
      if (src < SEQL) {
        z.x = Q[(rb + src) * DMODEL + d];
        z.y = Kv[(rb + src) * DMODEL + d];
      }
      buf[j] = z;
    }
    __syncthreads();
    for (int s = 1; s <= 12; ++s) {             // radix-2 DIT
      int mh = 1 << (s - 1);
      float base = -3.14159265358979323846f / (float)mh;
      for (int u = tid; u < 2048; u += 256) {
        int pos = u & (mh - 1);
        int grp = u >> (s - 1);
        int i0 = (grp << s) + pos, i1 = i0 + mh;
        float sn, cs; __sincosf(base * (float)pos, &sn, &cs);
        float2 wv = make_float2(cs, sn);
        float2 a = buf[i0];
        float2 b = cmul2(wv, buf[i1]);
        buf[i0] = make_float2(a.x + b.x, a.y + b.y);
        buf[i1] = make_float2(a.x - b.x, a.y - b.y);
      }
      __syncthreads();
    }
    // unpack q/k spectra, accumulate q * conj(k)
    for (int f = tid; f < NFFT; f += 256) {
      float2 z1 = buf[f], z2 = buf[(NFFT - f) & (NFFT - 1)];
      float qr = 0.5f * (z1.x + z2.x), qi = 0.5f * (z1.y - z2.y);
      float kr = 0.5f * (z1.y + z2.y), ki = 0.5f * (z2.x - z1.x);
      pac[f].x += qr * kr + qi * ki;
      pac[f].y += qi * kr - qr * ki;
    }
  }
  __syncthreads();
  float* dst = pgrp + ((size_t)bb * 16 + g) * NFFT * 2;
  for (int f = tid; f < NFFT; f += 256) {
    dst[f * 2 + 0] = pac[f].x;
    dst[f * 2 + 1] = pac[f].y;
  }
}

// ---------------------------------------------------------------------------
// Inverse FFT of summed cross-spectrum, circular corr recovery, top-7 +
// softmax weights.  grid (B), block 256.
// ---------------------------------------------------------------------------
__global__ void __launch_bounds__(256) corr_topk_kernel(
    const float* __restrict__ pgrp, float* __restrict__ wout,
    int* __restrict__ dout)
{
  __shared__ float2 buf[NFFT];
  __shared__ float corr[SEQL];
  __shared__ float rv[256];
  __shared__ int   ri[256];
  __shared__ float wts[8];
  __shared__ int   dls[8];
  int bb = blockIdx.x, tid = threadIdx.x;

  for (int j = tid; j < NFFT; j += 256) {       // bit-reversed load + group sum
    int src = __brev(j) >> 20;
    float xr = 0.f, xi = 0.f;
    for (int g = 0; g < 16; ++g) {
      const float* p = pgrp + (((size_t)bb * 16 + g) * NFFT + src) * 2;
      xr += p[0]; xi += p[1];
    }
    buf[j] = make_float2(xr, xi);
  }
  __syncthreads();
  for (int s = 1; s <= 12; ++s) {               // inverse: +angle twiddles
    int mh = 1 << (s - 1);
    float base = 3.14159265358979323846f / (float)mh;
    for (int u = tid; u < 2048; u += 256) {
      int pos = u & (mh - 1);
      int grp = u >> (s - 1);
      int i0 = (grp << s) + pos, i1 = i0 + mh;
      float sn, cs; __sincosf(base * (float)pos, &sn, &cs);
      float2 wv = make_float2(cs, sn);
      float2 a = buf[i0];
      float2 b = cmul2(wv, buf[i1]);
      buf[i0] = make_float2(a.x + b.x, a.y + b.y);
      buf[i1] = make_float2(a.x - b.x, a.y - b.y);
    }
    __syncthreads();
  }
  // circ corr of period L from 4096 linear corr; 1/N fft scale; 1/256 chan mean
  const float scale = 1.0f / (4096.0f * 256.0f);
  for (int t = tid; t < SEQL; t += 256)
    corr[t] = (buf[t].x + buf[t + 2560].x) * scale;
  __syncthreads();

  for (int it = 0; it < TOPK; ++it) {
    float best = -1e30f; int bi = 0;
    for (int t = tid; t < SEQL; t += 256)
      if (corr[t] > best) { best = corr[t]; bi = t; }
    rv[tid] = best; ri[tid] = bi;
    __syncthreads();
    if (tid == 0) {
      float bv = -1e30f; int bj = 0;
      for (int u = 0; u < 256; ++u)
        if (rv[u] > bv) { bv = rv[u]; bj = ri[u]; }
      wts[it] = bv; dls[it] = bj; corr[bj] = -1e30f;
    }
    __syncthreads();
  }
  if (tid == 0) {
    float mx = wts[0];
    for (int i = 1; i < TOPK; ++i) if (wts[i] > mx) mx = wts[i];
    float e[TOPK], ssum = 0.f;
    for (int i = 0; i < TOPK; ++i) { e[i] = __expf(wts[i] - mx); ssum += e[i]; }
    for (int i = 0; i < TOPK; ++i) {
      wout[bb * 8 + i] = e[i] / ssum;
      dout[bb * 8 + i] = dls[i];
    }
  }
}

// ---------------------------------------------------------------------------
// Aggregation: out[b,t,d] = sum_i w_i * V[b, (t+delay_i)%L, d]
// ---------------------------------------------------------------------------
__global__ void __launch_bounds__(256) agg_kernel(
    const float* __restrict__ V, const float* __restrict__ w,
    const int* __restrict__ dl, float* __restrict__ out)
{
  int t = blockIdx.x, bb = blockIdx.y, d = threadIdx.x;
  size_t rb = (size_t)bb * SEQL;
  float s = 0.f;
  #pragma unroll
  for (int i = 0; i < TOPK; ++i) {
    int tt = t + dl[bb * 8 + i];
    if (tt >= SEQL) tt -= SEQL;
    s += w[bb * 8 + i] * V[(rb + tt) * DMODEL + d];
  }
  out[(rb + t) * DMODEL + d] = s;
}

// ---------------------------------------------------------------------------
// Series decomposition: out = in - movavg_25(in) with replicate padding
// ---------------------------------------------------------------------------
__global__ void __launch_bounds__(256) decomp_kernel(
    const float* __restrict__ in, float* __restrict__ out)
{
  int t = blockIdx.x, bb = blockIdx.y, d = threadIdx.x;
  size_t rb = (size_t)bb * SEQL;
  float s = 0.f;
  #pragma unroll
  for (int j = -12; j <= 12; ++j) {
    int tc = t + j;
    tc = tc < 0 ? 0 : (tc > SEQL - 1 ? SEQL - 1 : tc);
    s += in[(rb + tc) * DMODEL + d];
  }
  out[(rb + t) * DMODEL + d] = in[(rb + t) * DMODEL + d] - s * (1.0f / 25.0f);
}

// ---------------------------------------------------------------------------
// LayerNorm over channels. grid (L, B), block 256 (=d)
// ---------------------------------------------------------------------------
__global__ void __launch_bounds__(256) ln_kernel(
    const float* __restrict__ X, const float* __restrict__ w,
    const float* __restrict__ b, float* __restrict__ O)
{
  __shared__ float red[256];
  int t = blockIdx.x, bb = blockIdx.y, d = threadIdx.x;
  size_t base = ((size_t)bb * SEQL + t) * DMODEL;
  float v = X[base + d];
  red[d] = v; __syncthreads();
  for (int s = 128; s > 0; s >>= 1) { if (d < s) red[d] += red[d + s]; __syncthreads(); }
  float mu = red[0] * (1.0f / 256.0f);
  __syncthreads();
  float c = v - mu;
  red[d] = c * c; __syncthreads();
  for (int s = 128; s > 0; s >>= 1) { if (d < s) red[d] += red[d + s]; __syncthreads(); }
  float var = red[0] * (1.0f / 256.0f);
  O[base + d] = c * rsqrtf(var + 1e-5f) * w[d] + b[d];
}

// ---------------------------------------------------------------------------
// Seq-mean subtract + mean/max pool + GELU.  grid (B), block 256 (=d)
// ---------------------------------------------------------------------------
__global__ void __launch_bounds__(256) colstat_kernel(
    const float* __restrict__ Xh, float* __restrict__ feat)
{
  int bb = blockIdx.x, d = threadIdx.x;
  float s = 0.f, m = -1e30f;
  for (int t = 0; t < SEQL; ++t) {
    float v = Xh[((size_t)bb * SEQL + t) * DMODEL + d];
    s += v; if (v > m) m = v;
  }
  float mu = s * (1.0f / (float)SEQL);
  feat[bb * 512 + d]       = gelu_f(s * (1.0f / (float)SEQL) - mu); // == gelu(0)
  feat[bb * 512 + 256 + d] = gelu_f(m - mu);
}

// ---------------------------------------------------------------------------
// Final projection [64,512] @ [176,512]^T + b.  grid (B), block 192.
// ---------------------------------------------------------------------------
__global__ void __launch_bounds__(192) proj_kernel(
    const float* __restrict__ feat, const float* __restrict__ W,
    const float* __restrict__ bias, float* __restrict__ out)
{
  int bb = blockIdx.x, o = threadIdx.x;
  if (o < 176) {
    float s = bias[o];
    const float* f = feat + bb * 512;
    const float* w = W + o * 512;
    for (int j = 0; j < 512; ++j) s += f[j] * w[j];
    out[bb * 176 + o] = s;
  }
}

// ---------------------------------------------------------------------------
extern "C" void kernel_launch(void* const* d_in, const int* in_sizes, int n_in,
                              void* d_out, int out_size, void* d_ws, size_t ws_size,
                              hipStream_t stream) {
  (void)in_sizes; (void)n_in; (void)out_size; (void)ws_size;
  const float* x_enc  = (const float*)d_in[0];
  const float* emb_w  = (const float*)d_in[1];
  const float* Wq     = (const float*)d_in[2];
  const float* bq     = (const float*)d_in[3];
  const float* Wk     = (const float*)d_in[4];
  const float* bk     = (const float*)d_in[5];
  const float* Wv     = (const float*)d_in[6];
  const float* bv     = (const float*)d_in[7];
  const float* Wo     = (const float*)d_in[8];
  const float* bo     = (const float*)d_in[9];
  const float* Wc1    = (const float*)d_in[10];
  const float* bc1    = (const float*)d_in[11];
  const float* Wc2    = (const float*)d_in[12];
  const float* bc2    = (const float*)d_in[13];
  const float* ln_w   = (const float*)d_in[14];
  const float* ln_b   = (const float*)d_in[15];
  const float* proj_w = (const float*)d_in[16];
  const float* proj_b = (const float*)d_in[17];

  // ---- workspace layout (floats), everything 16B-aligned ----
  const size_t SZ = (size_t)BATCHN * SEQL * DMODEL;   // 25,165,824
  float* ws   = (float*)d_ws;
  float* X    = ws;            // activations
  float* B1   = X  + SZ;       // q / agg / ffn-out
  float* B2   = B1 + SZ;       // k / attn-out
  float* B3   = B2 + SZ;       // v / ffn hidden chunk
  float* pgrp = B3 + SZ;                       // 64*16*4096*2 = 8,388,608
  float* wts  = pgrp + (size_t)BATCHN * 16 * NFFT * 2;  // 64*8
  int*   dls  = (int*)(wts + BATCHN * 8);               // 64*8
  float* feat = (float*)(dls + BATCHN * 8);             // 64*512
  _Float16* hW   = (_Float16*)(feat + BATCHN * 512);
  _Float16* hWq  = hW;                 // 2*256*256
  _Float16* hWk  = hWq  + 131072;
  _Float16* hWv  = hWk  + 131072;
  _Float16* hWo  = hWv  + 131072;
  _Float16* hWc1 = hWo  + 131072;      // 2*1024*256
  _Float16* hWc2 = hWc1 + 524288;      // 2*256*1024

  const dim3 blk(256);
  const int M = BATCHN * SEQL;         // 98304

  // weight conversion f32 -> f16 (WMMA operand precision)
  cvt_f16_kernel<<<dim3(131072 / 256), blk, 0, stream>>>(Wq,  hWq,  131072);
  cvt_f16_kernel<<<dim3(131072 / 256), blk, 0, stream>>>(Wk,  hWk,  131072);
  cvt_f16_kernel<<<dim3(131072 / 256), blk, 0, stream>>>(Wv,  hWv,  131072);
  cvt_f16_kernel<<<dim3(131072 / 256), blk, 0, stream>>>(Wo,  hWo,  131072);
  cvt_f16_kernel<<<dim3(524288 / 256), blk, 0, stream>>>(Wc1, hWc1, 524288);
  cvt_f16_kernel<<<dim3(524288 / 256), blk, 0, stream>>>(Wc2, hWc2, 524288);

  // token embedding
  embed_kernel<<<dim3(SEQL, BATCHN), blk, 0, stream>>>(x_enc, emb_w, X);

  for (int l = 0; l < 2; ++l) {
    const _Float16* wq  = hWq  + (size_t)l * 65536;
    const _Float16* wk  = hWk  + (size_t)l * 65536;
    const _Float16* wv  = hWv  + (size_t)l * 65536;
    const _Float16* wo  = hWo  + (size_t)l * 65536;
    const _Float16* wc1 = hWc1 + (size_t)l * 262144;
    const _Float16* wc2 = hWc2 + (size_t)l * 262144;

    // Q, K, V projections (WMMA GEMMs)
    gemm_wmma_kernel<0><<<dim3(M / 128, 4), blk, 0, stream>>>(
        X, wq, bq + l * 256, nullptr, B1, M, 256, 256);
    gemm_wmma_kernel<0><<<dim3(M / 128, 4), blk, 0, stream>>>(
        X, wk, bk + l * 256, nullptr, B2, M, 256, 256);
    gemm_wmma_kernel<0><<<dim3(M / 128, 4), blk, 0, stream>>>(
        X, wv, bv + l * 256, nullptr, B3, M, 256, 256);

    // channel-mean autocorrelation via packed 4096-pt FFT
    fft_qk_kernel<<<dim3(16, BATCHN), blk, 0, stream>>>(B1, B2, pgrp);
    corr_topk_kernel<<<dim3(BATCHN), blk, 0, stream>>>(pgrp, wts, dls);

    // delay aggregation of V (q buffer reused)
    agg_kernel<<<dim3(SEQL, BATCHN), blk, 0, stream>>>(B3, wts, dls, B1);

    // output projection + residual, then decomposition
    gemm_wmma_kernel<2><<<dim3(M / 128, 4), blk, 0, stream>>>(
        B1, wo, bo + l * 256, X, B2, M, 256, 256);
    decomp_kernel<<<dim3(SEQL, BATCHN), blk, 0, stream>>>(B2, X);

    // FFN, chunked over 6 x 16384 rows so hidden fits in B3
    for (int c = 0; c < 6; ++c) {
      const float* Ax = X  + (size_t)c * 16384 * 256;
      float*       Cx = B1 + (size_t)c * 16384 * 256;
      gemm_wmma_kernel<1><<<dim3(128, 16), blk, 0, stream>>>(
          Ax, wc1, bc1 + l * 1024, nullptr, B3, 16384, 1024, 256);
      gemm_wmma_kernel<2><<<dim3(128, 4), blk, 0, stream>>>(
          B3, wc2, bc2 + l * 256, Ax, Cx, 16384, 256, 1024);
    }
    decomp_kernel<<<dim3(SEQL, BATCHN), blk, 0, stream>>>(B1, X);
  }

  // final norm + pooling + projection
  ln_kernel<<<dim3(SEQL, BATCHN), blk, 0, stream>>>(X, ln_w, ln_b, B1);
  colstat_kernel<<<dim3(BATCHN), blk, 0, stream>>>(B1, feat);
  proj_kernel<<<dim3(BATCHN), dim3(192), 0, stream>>>(
      feat, proj_w, proj_b, (float*)d_out);
}
